// STGCNBlock_82240033784096
// MI455X (gfx1250) — matrix operand order repaired and checked by reading, not compiled, not run on hardware
//
#include <hip/hip_runtime.h>
#include <cstdint>

// ---------------------------------------------------------------------------
// STGCN block for MI455X (gfx1250, wave32, WMMA + TDM + async-LDS staging).
//  - GEMM stages: v_wmma_f32_16x16x32_bf16, fragments as b128 loads
//  - conv weights double-buffered into LDS by the Tensor Data Mover
//  - GAT linear weights async-staged to LDS (ASYNCcnt path)
//  - graph stages: f32 VALU + L2-resident atomics
// ---------------------------------------------------------------------------

typedef __attribute__((ext_vector_type(16))) __bf16    bf16x16;
typedef __attribute__((ext_vector_type(8)))  float     f32x8;
typedef __attribute__((ext_vector_type(4)))  int       i32x4;
typedef __attribute__((ext_vector_type(8)))  int       i32x8;
typedef __attribute__((ext_vector_type(4)))  unsigned  u32x4;

struct V8 { i32x4 lo, hi; };   // 32B carrier for one 16-element bf16 fragment

constexpr int Bc   = 128;           // batch
constexpr int CIN1 = 64;            // conv1 in-channels
constexpr int Tc   = 2048;          // time
constexpr int Hc   = 128;           // hidden
constexpr int KK   = 9;             // conv taps (SAME, pad 4)
constexpr int Nn   = Bc * Tc;       // 262144 graph nodes
constexpr int TILE_T = 24;          // 16 outputs + 8 halo (pad 4 each side)

static __device__ __forceinline__ unsigned short bfbits(float f) {
  unsigned u = __builtin_bit_cast(unsigned, f);
  u = u + 0x7FFFu + ((u >> 16) & 1u);                  // round-to-nearest-even
  return (unsigned short)(u >> 16);
}
// order-preserving float<->uint map for atomicMax-based segment_max
static __device__ __forceinline__ unsigned fenc(float f) {
  unsigned u = __builtin_bit_cast(unsigned, f);
  return (u & 0x80000000u) ? ~u : (u | 0x80000000u);
}
static __device__ __forceinline__ float fdec(unsigned e) {
  unsigned u = (e & 0x80000000u) ? (e & 0x7fffffffu) : ~e;
  return __builtin_bit_cast(float, u);
}

// CDNA5 async copy: 16B global -> LDS, tracked by ASYNCcnt (ISA 15.18.3)
static __device__ __forceinline__ void async_b128(unsigned lds_byte_off,
                                                  const unsigned short* g) {
  asm volatile("global_load_async_to_lds_b128 %0, %1, off"
               :: "v"(lds_byte_off), "v"(g) : "memory");
}
static __device__ __forceinline__ void wait_async0() {
  asm volatile("s_wait_asynccnt 0x0" ::: "memory");
}

// CDNA5 Tensor Data Mover: 1-D tile (n_dwords x 4B), global -> LDS.
// D# layout per ISA 8.3/8.4: count=1, type=2, data_size=4B, dim1=tile1=1.
// This toolchain exposes the 6-arg builtin: (g0, g1, g2, g3, g4, cpol).
static __device__ __forceinline__ void tdm_load_1d(unsigned lds_byte_off,
                                                   const void* gsrc,
                                                   unsigned n_dwords) {
  const unsigned long long ga = (unsigned long long)gsrc;
  u32x4 g0;
  g0[0] = 1u;                                          // count=1 (valid user D#)
  g0[1] = lds_byte_off;                                // lds_addr
  g0[2] = (unsigned)ga;                                // global_addr[95:64]
  g0[3] = (unsigned)(ga >> 32) | (2u << 30);           // addr hi | type=2
  i32x8 g1;
  g1[0] = (int)(2u << 16);                             // data_size=2 (4 bytes)
  g1[1] = (int)((n_dwords & 0xffffu) << 16);           // tensor_dim0[15:0]
  g1[2] = (int)((n_dwords >> 16) | (1u << 16));        // dim0[31:16] | tensor_dim1=1
  g1[3] = (int)((n_dwords & 0xffffu) << 16);           // tile_dim0 (<=65535)
  g1[4] = 1;                                           // tile_dim1=1
  g1[5] = (int)n_dwords;                               // tensor_dim0_stride lo
  g1[6] = (int)((n_dwords & 0xffffu) << 16);           // stride0 hi=0 | stride1 lo
  g1[7] = (int)(n_dwords >> 16);                       // stride1 hi
  i32x4 z;
  z[0] = 0; z[1] = 0; z[2] = 0; z[3] = 0;              // groups 2/3 unused (2-D max)
  i32x8 z8;
  #pragma unroll
  for (int i = 0; i < 8; ++i) z8[i] = 0;
  __builtin_amdgcn_tensor_load_to_lds(g0, g1, z, z, z8, 0);
}

// ------------------------ precision / layout pre-passes --------------------
__global__ __launch_bounds__(256)
void cvt_bf16(const float* __restrict__ src, unsigned short* __restrict__ dst, int n) {
  const int stride = gridDim.x * blockDim.x;
  for (int i = blockIdx.x * blockDim.x + threadIdx.x; i < n; i += stride)
    dst[i] = bfbits(src[i]);
}

// conv weights [H][CI][9] (OIH) -> bf16 [kt][H][CI] (tap-major, pairs contiguous)
__global__ __launch_bounds__(256)
void rearrange_w(const float* __restrict__ w, unsigned short* __restrict__ dst,
                 int HO, int CI) {
  const int n = HO * CI * KK;
  const int stride = gridDim.x * blockDim.x;
  for (int i = blockIdx.x * blockDim.x + threadIdx.x; i < n; i += stride) {
    const int h   = i / (CI * KK);
    const int rem = i - h * CI * KK;
    const int c   = rem / KK;
    const int kt  = rem - c * KK;
    dst[(size_t)kt * HO * CI + h * CI + c] = bfbits(w[i]);
  }
}

// ---------------------------------------------------------------------------
// Implicit-GEMM 1D conv (SAME, K=9), bf16 in / f32 accumulate.
// One wave = 16(t) x 16(h) tile; 8 waves/block cover all 128 channels.
// Weights: per-tap [H][CIN_] slice double-buffered in LDS by the TDM.
// Activations: [TILE_T][CIN_] transposed tile in LDS (halo zero-padded).
// Fragments = two b128 loads each (16 contiguous k per lane half: k = 16*half + s).
// ---------------------------------------------------------------------------
template<int CIN_, bool OUTBF_>
__global__ __launch_bounds__(256)
void conv1d_wmma(const unsigned short* __restrict__ xbf,  // [B][CIN_][T] bf16
                 const unsigned short* __restrict__ wbf,  // [9][H][CIN_] bf16
                 const float* __restrict__ bias,          // [H]
                 unsigned short* __restrict__ obf,        // bf16 out (conv1)
                 float* __restrict__ of32)                // f32 out (conv2)
{
  __shared__ unsigned short smem[2 * Hc * CIN_ + TILE_T * CIN_];
  unsigned short* wlds = smem;                    // 2 x [H][CIN_]
  unsigned short* alds = smem + 2 * Hc * CIN_;    // [TILE_T][CIN_]

  const int lane = threadIdx.x & 31;
  const int wv   = threadIdx.x >> 5;
  const int t0   = blockIdx.x * 16;
  const int b    = blockIdx.y;
  const int h0   = wv * 16;
  const int ln   = lane & 15;     // A row (t) and B col (h)
  const int grp  = lane >> 4;

  constexpr unsigned WSLICE_DW = (unsigned)(Hc * CIN_ / 2);   // dwords per tap

  // ---- stage activation tile (transposed, zero halo), scalar ds stores ----
  for (int i = threadIdx.x; i < CIN_ * TILE_T; i += 256) {
    const int c    = i / TILE_T;
    const int tloc = i - c * TILE_T;
    const int t    = t0 + tloc - 4;
    unsigned short v = 0;
    if (t >= 0 && t < Tc) v = xbf[((size_t)b * CIN_ + c) * Tc + t];
    alds[tloc * CIN_ + c] = v;
  }

  // ---- TDM-stage tap 0 weights into buffer 0 (wave 0 issues) ----
  const unsigned wlds_base = (unsigned)(size_t)wlds;
  if (wv == 0) tdm_load_1d(wlds_base, wbf, WSLICE_DW);

  f32x8 acc;
  #pragma unroll
  for (int i = 0; i < 8; ++i) acc[i] = 0.f;

  for (int kt = 0; kt < KK; ++kt) {
    __builtin_amdgcn_s_wait_tensorcnt(0);
    __syncthreads();                              // weights+acts visible

    if (wv == 0 && kt + 1 < KK)                   // TDM-prefetch next tap
      tdm_load_1d(wlds_base + ((kt + 1) & 1) * (Hc * CIN_ * 2),
                  wbf + (size_t)(kt + 1) * Hc * CIN_, WSLICE_DW);

    const unsigned short* wbuf = wlds + (kt & 1) * (Hc * CIN_);
    const int tloc = ln + kt;                     // t = t0 + ln + kt - 4
    #pragma unroll
    for (int c0 = 0; c0 < CIN_; c0 += 32) {
      // A 16x32 bf16 fragment: dwords contiguous per half (k = 16*half + s)
      const unsigned short* ap = alds + tloc * CIN_ + c0 + 16 * grp;
      const V8 av = { *(const i32x4*)ap, *(const i32x4*)(ap + 8) };
      // B 32x16 bf16 fragment: 8 contiguous dwords per lane
      const unsigned short* bp = wbuf + (h0 + ln) * CIN_ + c0 + 16 * grp;
      const V8 bv = { *(const i32x4*)bp, *(const i32x4*)(bp + 8) };
      acc = __builtin_amdgcn_wmma_f32_16x16x32_bf16(
          false, __builtin_bit_cast(bf16x16, av),
          false, __builtin_bit_cast(bf16x16, bv),
          (short)0, acc, false, false);
    }
  }

  const float bvs = bias[h0 + ln];
  #pragma unroll
  for (int r = 0; r < 8; ++r) {                   // C/D: row = r + 8*grp, col = ln
    const int    tt = t0 + r + 8 * grp;
    float        v  = acc[r] + bvs;
    v = v > 0.f ? v : 0.f;                        // fused ReLU
    const size_t o = ((size_t)b * Hc + (h0 + ln)) * Tc + tt;
    if (OUTBF_) obf[o]  = bfbits(v);
    else        of32[o] = v;
  }
}

// ---------------------------------------------------------------------------
// Dual GEMM: xl = nf @ Wl^T + bl ; xr = nf @ Wr^T + br.  All-bf16 inputs.
// Wl/Wr (64KB) async-staged to LDS once per block (ASYNCcnt path); A from
// global as b128, B fragments as ds_load_b128.
// ---------------------------------------------------------------------------
__global__ __launch_bounds__(256)
void gat_lin_wmma(const unsigned short* __restrict__ nf,   // [N][128] bf16
                  const unsigned short* __restrict__ Wlb,  // [128][128] bf16
                  const float* __restrict__ bl,
                  const unsigned short* __restrict__ Wrb,
                  const float* __restrict__ br,
                  float* __restrict__ xl, float* __restrict__ xr)
{
  __shared__ unsigned short wsm[2 * Hc * Hc];     // Wl then Wr, 64KB
  const int lane = threadIdx.x & 31;
  const int wv   = threadIdx.x >> 5;
  const int row0 = blockIdx.x * 16;
  const int h0   = wv * 16;
  const int ln   = lane & 15;
  const int grp  = lane >> 4;

  // async-stage both weight matrices (2048+2048 b128 chunks over 256 threads)
  const unsigned base = (unsigned)(size_t)wsm;
  #pragma unroll
  for (int i = 0; i < 8; ++i) {
    const int chunk = threadIdx.x + i * 256;      // 16B chunks
    async_b128(base + chunk * 16u, Wlb + chunk * 8);
    async_b128(base + (Hc * Hc * 2) + chunk * 16u, Wrb + chunk * 8);
  }
  wait_async0();
  __syncthreads();

  f32x8 aL, aR;
  #pragma unroll
  for (int i = 0; i < 8; ++i) { aL[i] = 0.f; aR[i] = 0.f; }

  #pragma unroll
  for (int c0 = 0; c0 < Hc; c0 += 32) {
    const unsigned short* ap = nf + (size_t)(row0 + ln) * Hc + c0 + 16 * grp;
    const V8 av = { *(const i32x4*)ap, *(const i32x4*)(ap + 8) };
    const unsigned short* lp = wsm + (h0 + ln) * Hc + c0 + 16 * grp;
    const V8 lv = { *(const i32x4*)lp, *(const i32x4*)(lp + 8) };
    const unsigned short* rp = wsm + Hc * Hc + (h0 + ln) * Hc + c0 + 16 * grp;
    const V8 rv = { *(const i32x4*)rp, *(const i32x4*)(rp + 8) };
    const bf16x16 A = __builtin_bit_cast(bf16x16, av);
    aL = __builtin_amdgcn_wmma_f32_16x16x32_bf16(
        false, A, false, __builtin_bit_cast(bf16x16, lv), (short)0, aL, false, false);
    aR = __builtin_amdgcn_wmma_f32_16x16x32_bf16(
        false, A, false, __builtin_bit_cast(bf16x16, rv), (short)0, aR, false, false);
  }
  const float bLv = bl[h0 + ln], bRv = br[h0 + ln];
  #pragma unroll
  for (int r = 0; r < 8; ++r) {
    const size_t row = (size_t)(row0 + r + 8 * grp);
    xl[row * Hc + h0 + ln] = aL[r] + bLv;
    xr[row * Hc + h0 + ln] = aR[r] + bRv;
  }
}

// --------------------------- edge kernels ----------------------------------
__global__ __launch_bounds__(256)
void edge_score(const long long* __restrict__ ei, int E, int Etot,
                const float* __restrict__ xl, const float* __restrict__ xr,
                const float* __restrict__ att, float* __restrict__ esc,
                unsigned* __restrict__ menc)
{
  const int lane = threadIdx.x & 31, wv = threadIdx.x >> 5;
  const int e = blockIdx.x * 8 + wv;
  if (e >= Etot) return;
  int src, dst;
  if (e < E) { src = (int)ei[e]; dst = (int)ei[(size_t)E + e]; }
  else       { src = dst = e - E; }                                // self loop
  const int c = lane * 4;
  const float4 vl = *(const float4*)&xl[(size_t)src * Hc + c];
  const float4 vr = *(const float4*)&xr[(size_t)dst * Hc + c];
  const float4 va = *(const float4*)&att[c];
  float s = 0.f, t;
  t = vl.x + vr.x; t = t > 0.f ? t : 0.2f * t; s += t * va.x;
  t = vl.y + vr.y; t = t > 0.f ? t : 0.2f * t; s += t * va.y;
  t = vl.z + vr.z; t = t > 0.f ? t : 0.2f * t; s += t * va.z;
  t = vl.w + vr.w; t = t > 0.f ? t : 0.2f * t; s += t * va.w;
  #pragma unroll
  for (int off = 16; off; off >>= 1) s += __shfl_xor(s, off, 32);
  if (lane == 0) { esc[e] = s; atomicMax(&menc[dst], fenc(s)); }
}

__global__ __launch_bounds__(256)
void edge_exp(const long long* __restrict__ ei, int E, int Etot,
              float* __restrict__ esc, const unsigned* __restrict__ menc,
              float* __restrict__ denom)
{
  const int e = blockIdx.x * blockDim.x + threadIdx.x;
  if (e >= Etot) return;
  const int dst = (e < E) ? (int)ei[(size_t)E + e] : (e - E);
  const float a = __expf(esc[e] - fdec(menc[dst]));
  esc[e] = a;
  atomicAdd(&denom[dst], a);
}

__global__ __launch_bounds__(256)
void edge_scatter(const long long* __restrict__ ei, int E, int Etot,
                  const float* __restrict__ esc, const float* __restrict__ denom,
                  const float* __restrict__ xl, float* __restrict__ gat)
{
  const int lane = threadIdx.x & 31, wv = threadIdx.x >> 5;
  const int e = blockIdx.x * 8 + wv;
  if (e >= Etot) return;
  int src, dst;
  if (e < E) { src = (int)ei[e]; dst = (int)ei[(size_t)E + e]; }
  else       { src = dst = e - E; }
  const float alpha = esc[e] / (denom[dst] + 1e-16f);
  const int c = lane * 4;
  const float4 v = *(const float4*)&xl[(size_t)src * Hc + c];
  float* g = &gat[(size_t)dst * Hc + c];
  atomicAdd(g + 0, alpha * v.x);
  atomicAdd(g + 1, alpha * v.y);
  atomicAdd(g + 2, alpha * v.z);
  atomicAdd(g + 3, alpha * v.w);
}

// ---------------- GAT bias + ReLU + BN batch statistics --------------------
__global__ __launch_bounds__(256)
void gat_bias_relu_stats(float* __restrict__ gat, const float* __restrict__ gbias,
                         float* __restrict__ sums, float* __restrict__ sumsq)
{
  const size_t stride = (size_t)gridDim.x * blockDim.x;   // multiple of 128
  size_t i = (size_t)blockIdx.x * blockDim.x + threadIdx.x;
  const int   c  = (int)(i & (size_t)(Hc - 1));
  const float bv = gbias[c];
  float s = 0.f, q = 0.f;
  for (; i < (size_t)Nn * Hc; i += stride) {
    float v = gat[i] + bv;
    v = v > 0.f ? v : 0.f;
    gat[i] = v;
    s += v; q += v * v;
  }
  atomicAdd(&sums[c], s);
  atomicAdd(&sumsq[c], q);
}

__global__ void bn_finalize(const float* __restrict__ sums, const float* __restrict__ sumsq,
                            const float* __restrict__ gamma, const float* __restrict__ beta,
                            float* __restrict__ scale, float* __restrict__ shift)
{
  const int c = threadIdx.x;
  const float inv = 1.0f / (float)Nn;
  const float mu  = sums[c] * inv;
  const float var = sumsq[c] * inv - mu * mu;             // biased var
  const float sc  = gamma[c] * rsqrtf(var + 1e-5f);
  scale[c] = sc;
  shift[c] = beta[c] - mu * sc;
}

// normalize + convert to bf16 for conv2's A path
__global__ __launch_bounds__(256)
void bn_apply_cvt(const float* __restrict__ gat,
                  const float* __restrict__ scale, const float* __restrict__ shift,
                  unsigned short* __restrict__ out)
{
  const size_t stride = (size_t)gridDim.x * blockDim.x;   // multiple of 128
  for (size_t i = (size_t)blockIdx.x * blockDim.x + threadIdx.x;
       i < (size_t)Nn * Hc; i += stride) {
    const int c = (int)(i & (size_t)(Hc - 1));
    out[i] = bfbits(gat[i] * scale[c] + shift[c]);
  }
}

// ---------------------------------------------------------------------------
extern "C" void kernel_launch(void* const* d_in, const int* in_sizes, int n_in,
                              void* d_out, int out_size, void* d_ws, size_t ws_size,
                              hipStream_t stream)
{
  (void)n_in; (void)out_size; (void)ws_size;
  const float*     x        = (const float*)d_in[0];
  const long long* ei       = (const long long*)d_in[1];   // int64 per reference
  const float*     conv1_w  = (const float*)d_in[2];
  const float*     conv1_b  = (const float*)d_in[3];
  const float*     Wl       = (const float*)d_in[4];
  const float*     bl       = (const float*)d_in[5];
  const float*     Wr       = (const float*)d_in[6];
  const float*     br       = (const float*)d_in[7];
  const float*     att      = (const float*)d_in[8];
  const float*     gat_bias = (const float*)d_in[9];
  const float*     bn_gamma = (const float*)d_in[10];
  const float*     bn_beta  = (const float*)d_in[11];
  const float*     conv2_w  = (const float*)d_in[12];
  const float*     conv2_b  = (const float*)d_in[13];
  float*           out      = (float*)d_out;

  const int E    = in_sizes[1] / 2;
  const int Etot = E + Nn;

  // workspace carve-out (256B aligned)
  char*  ws  = (char*)d_ws;
  size_t off = 0;
  auto take = [&](size_t bytes) -> char* {
    char* p = ws + off;
    off = (off + bytes + 255) & ~(size_t)255;
    return p;
  };
  unsigned short* nf    = (unsigned short*)take((size_t)Bc * Hc * Tc * 2); // bf16 feats (reused for conv2 input)
  unsigned short* xbf   = (unsigned short*)take((size_t)Bc * CIN1 * Tc * 2);
  unsigned short* w1r   = (unsigned short*)take((size_t)KK * Hc * CIN1 * 2);
  unsigned short* w2r   = (unsigned short*)take((size_t)KK * Hc * Hc * 2);
  unsigned short* wlbf  = (unsigned short*)take((size_t)Hc * Hc * 2);
  unsigned short* wrbf  = (unsigned short*)take((size_t)Hc * Hc * 2);
  float*          xl    = (float*)take((size_t)Nn * Hc * 4);
  float*          xr    = (float*)take((size_t)Nn * Hc * 4);
  float*          gat   = (float*)take((size_t)Nn * Hc * 4);
  float*          esc   = (float*)take((size_t)Etot * 4);
  unsigned*       menc  = (unsigned*)take((size_t)Nn * 4);
  float*          denom = (float*)take((size_t)Nn * 4);
  float*          sums  = (float*)take(Hc * 4);
  float*          sumsq = (float*)take(Hc * 4);
  float*          scale = (float*)take(Hc * 4);
  float*          shift = (float*)take(Hc * 4);

  // deterministic per-call init (graph-capture-safe async memsets)
  (void)hipMemsetAsync(gat,   0, (size_t)Nn * Hc * 4, stream);
  (void)hipMemsetAsync(menc,  0, (size_t)Nn * 4, stream);   // 0 == encoded -max
  (void)hipMemsetAsync(denom, 0, (size_t)Nn * 4, stream);
  (void)hipMemsetAsync(sums,  0, Hc * 4, stream);
  (void)hipMemsetAsync(sumsq, 0, Hc * 4, stream);

  // 0) one-time precision/layout passes
  cvt_bf16<<<2048, 256, 0, stream>>>(x, xbf, Bc * CIN1 * Tc);
  rearrange_w<<<128, 256, 0, stream>>>(conv1_w, w1r, Hc, CIN1);
  rearrange_w<<<256, 256, 0, stream>>>(conv2_w, w2r, Hc, Hc);
  cvt_bf16<<<64, 256, 0, stream>>>(Wl, wlbf, Hc * Hc);
  cvt_bf16<<<64, 256, 0, stream>>>(Wr, wrbf, Hc * Hc);

  const dim3 cg(Tc / 16, Bc);   // 128 x 128 blocks, 256 thr (8 waves -> 128 h)

  // 1) conv1 + ReLU -> bf16 node features (flat view == [N,128] row-major)
  conv1d_wmma<CIN1, true><<<cg, 256, 0, stream>>>(xbf, w1r, conv1_b, nf, nullptr);

  // 2) xl / xr linears (shared A fragment, LDS-resident weights)
  gat_lin_wmma<<<Nn / 16, 256, 0, stream>>>(nf, wlbf, bl, wrbf, br, xl, xr);

  // 3) GATv2 attention: scores + segment max
  edge_score<<<(Etot + 7) / 8, 256, 0, stream>>>(ei, E, Etot, xl, xr, att, esc, menc);

  // 4) exp + segment sum
  edge_exp<<<(Etot + 255) / 256, 256, 0, stream>>>(ei, E, Etot, esc, menc, denom);

  // 5) weighted scatter-add of xl[src] into gat[dst]
  edge_scatter<<<(Etot + 7) / 8, 256, 0, stream>>>(ei, E, Etot, esc, denom, xl, gat);

  // 6) + bias, ReLU, BN batch stats
  gat_bias_relu_stats<<<1024, 256, 0, stream>>>(gat, gat_bias, sums, sumsq);

  // 7) fold BN into per-channel scale/shift
  bn_finalize<<<1, Hc, 0, stream>>>(sums, sumsq, bn_gamma, bn_beta, scale, shift);

  // 8) BN-normalize + bf16 (conv2 A input; reuses nf buffer)
  bn_apply_cvt<<<1024, 256, 0, stream>>>(gat, scale, shift, nf);

  // 9) conv2 + ReLU -> d_out [B,H,T]
  conv1d_wmma<Hc, false><<<cg, 256, 0, stream>>>(nf, w2r, conv2_b, nullptr, out);
}
